// kemeny_transformer_80376017977631
// MI455X (gfx1250) — matrix operand order repaired and checked by reading, not compiled, not run on hardware
//
#include <hip/hip_runtime.h>

// MI455X (gfx1250) multi-head attention with tanh logit clipping.
// b=16, sq=skv=1024, emb=1024, heads=16, dh=64. fp32 I/O, bf16 WMMA math.
//
// Outputs (concatenated in d_out):
//   [0)              attn_output  (16,1024,1024) fp32
//   [16M)            attn_weights (16,1024,1024) fp32  (mean of P over heads)

typedef __attribute__((ext_vector_type(16))) __bf16 v16bf;
typedef __attribute__((ext_vector_type(8)))  float  v8f;

#define BSZ   16
#define SQ    1024
#define SKV   1024
#define EMB   1024
#define NH    16
#define DH    64
#define QT    16      // q rows per block (one WMMA M tile)
#define NWAVE 8       // 256 threads = 8 wave32
#define SCALE 0.125f  // 1/sqrt(64)
#define NEGINF (-1e9f)

// 10*tanh(x), overflow-safe, using fast v_rcp_f32 instead of IEEE divide.
// bf16 inputs make 1-ulp reciprocal error irrelevant.
__device__ __forceinline__ float tanh_clip10(float x) {
  float xc = fminf(fmaxf(x, -20.f), 20.f);
  float e  = __expf(2.f * xc);
  return 10.f * (e - 1.f) * __builtin_amdgcn_rcpf(e + 1.f);
}

__launch_bounds__(256)
__global__ void attn_tanhclip_kernel(const float* __restrict__ Q,
                                     const float* __restrict__ K,
                                     const float* __restrict__ V,
                                     const unsigned char* __restrict__ mask, // jax bool = 1 byte
                                     float* __restrict__ outO,
                                     float* __restrict__ outW) {
  __shared__ float S_lds[QT][SKV];  // 64 KB: scores -> probabilities (per head)
  __shared__ float O_lds[QT][DH];   //  4 KB: per-head output accumulator

  const int tid  = threadIdx.x;
  const int wave = tid >> 5;
  const int lane = tid & 31;
  const int half = lane >> 4;   // WMMA lane-half
  const int l16  = lane & 15;

  const int qtile = blockIdx.x;        // 0..63
  const int b     = blockIdx.y;        // 0..15
  const int q0    = qtile * QT;

  const unsigned char* mrow = mask + (size_t)b * SKV;

  // Head-mean P accumulator, register-resident.
  // This thread owns rows r = 2*wave + rr, cols c = lane + 32*j.
  float w_acc[2][32];
#pragma unroll
  for (int rr = 0; rr < 2; ++rr)
#pragma unroll
    for (int j = 0; j < 32; ++j) w_acc[rr][j] = 0.f;

  for (int h = 0; h < NH; ++h) {
    // ---- zero the per-head O accumulator ----
    for (int idx = tid; idx < QT * DH; idx += 256)
      ((float*)O_lds)[idx] = 0.f;

    // ---- Q A-fragments (16x32 bf16, two dh chunks), held in VGPRs for the head ----
    // A fragment (this code's contiguous k order, same as B): lane m = l16; a[i] <- k = half*16 + i
    const float* qrow = Q + ((size_t)(b * SQ + q0 + l16)) * EMB + h * DH;
    v16bf aQ0, aQ1;
#pragma unroll
    for (int i = 0; i < 16; ++i) {
      int k = half * 16 + i;
      aQ0[i] = (__bf16)qrow[k];
      aQ1[i] = (__bf16)qrow[32 + k];
    }
    __syncthreads();  // O zeroed; previous head's phase-4 reads complete

    // ---- Phase 1: S = mask( 10*tanh( (Q K^T) * scale ) ) ----
    // Each wave owns k-tiles {wave, wave+8, ...} of 64.
    for (int kt = wave; kt < SKV / 16; kt += NWAVE) {
      // B fragment (this code's contiguous k order, same as A): lane n = l16; b[i] <- kk = half*16 + i
      // B = K^T, so column n = contiguous dh slice of K row (kv0+n).
      const float* krow =
          K + ((size_t)(b * SKV + kt * 16 + l16)) * EMB + h * DH + half * 16;
      v16bf bK0, bK1;
#pragma unroll
      for (int i = 0; i < 16; ++i) {
        bK0[i] = (__bf16)krow[i];
        bK1[i] = (__bf16)krow[32 + i];
      }
      v8f c = {};
      c = __builtin_amdgcn_wmma_f32_16x16x32_bf16(false, aQ0, false, bK0,
                                                  (short)0, c, false, false);
      c = __builtin_amdgcn_wmma_f32_16x16x32_bf16(false, aQ1, false, bK1,
                                                  (short)0, c, false, false);
      const int col = kt * 16 + l16;
      // Branch-free masking: compute tanh unconditionally (cheap now), select.
      const float mfin = (mrow[col] != 0) ? NEGINF : 0.f;
#pragma unroll
      for (int v = 0; v < 8; ++v) {
        const int m = v + 8 * half;  // C/D layout: lanes 16-31 hold M+8
        float s = tanh_clip10(c[v] * SCALE);
        S_lds[m][col] = (mfin != 0.f) ? mfin : s;  // v_cndmask, no exec branching
      }
    }
    __syncthreads();

    // ---- Phase 2: row softmax (fp32), accumulate head-mean into registers ----
#pragma unroll
    for (int rr = 0; rr < 2; ++rr) {
      const int r = 2 * wave + rr;
      float mx = NEGINF;
#pragma unroll
      for (int j = 0; j < 32; ++j) mx = fmaxf(mx, S_lds[r][lane + 32 * j]);
#pragma unroll
      for (int off = 16; off > 0; off >>= 1) mx = fmaxf(mx, __shfl_xor(mx, off, 32));
      float sum = 0.f;
#pragma unroll
      for (int j = 0; j < 32; ++j) {
        float e = __expf(S_lds[r][lane + 32 * j] - mx);
        S_lds[r][lane + 32 * j] = e;
        sum += e;
      }
#pragma unroll
      for (int off = 16; off > 0; off >>= 1) sum += __shfl_xor(sum, off, 32);
      const float inv = __builtin_amdgcn_rcpf(sum);  // fast v_rcp_f32
#pragma unroll
      for (int j = 0; j < 32; ++j) {
        float p = S_lds[r][lane + 32 * j] * inv;
        S_lds[r][lane + 32 * j] = p;                 // keep P for the O GEMM
        w_acc[rr][j] += p * (1.f / (float)NH);       // head mean
      }
    }
    __syncthreads();

    // ---- Phase 3: O = P V. Each wave owns kv-chunks {wave, wave+8, ...} of 32. ----
    v8f o[4] = {v8f{}, v8f{}, v8f{}, v8f{}};
    for (int ch = wave; ch < SKV / 32; ch += NWAVE) {
      v16bf aP;  // P as A fragment (16x32 over kv)
#pragma unroll
      for (int i = 0; i < 16; ++i) {
        int k = half * 16 + i;
        aP[i] = (__bf16)S_lds[l16][ch * 32 + k];
      }
      const float* vbase =
          V + ((size_t)(b * SKV + ch * 32 + half * 16)) * EMB + h * DH + l16;
#pragma unroll
      for (int nt = 0; nt < 4; ++nt) {  // dh = 4 N-tiles of 16
        v16bf bV;
#pragma unroll
        for (int i = 0; i < 16; ++i)
          bV[i] = (__bf16)vbase[(size_t)i * EMB + nt * 16];
        o[nt] = __builtin_amdgcn_wmma_f32_16x16x32_bf16(false, aP, false, bV,
                                                        (short)0, o[nt], false, false);
      }
    }
    // merge wave partials (kv reduction) via LDS float atomics (ds_add_f32)
#pragma unroll
    for (int nt = 0; nt < 4; ++nt)
#pragma unroll
      for (int v = 0; v < 8; ++v)
        atomicAdd(&O_lds[v + 8 * half][nt * 16 + l16], o[nt][v]);
    __syncthreads();

    // ---- Phase 4: write this head's output slice ----
    float* obase = outO + ((size_t)(b * SQ + q0)) * EMB + h * DH;
    for (int idx = tid; idx < QT * DH; idx += 256) {
      int m = idx >> 6, d = idx & 63;
      obase[(size_t)m * EMB + d] = O_lds[m][d];
    }
    __syncthreads();
  }

  // ---- write head-averaged attention weights ----
  float* wbase = outW + ((size_t)(b * SQ + q0)) * SKV;
#pragma unroll
  for (int rr = 0; rr < 2; ++rr) {
    const int r = 2 * wave + rr;
#pragma unroll
    for (int j = 0; j < 32; ++j)
      wbase[(size_t)r * SKV + lane + 32 * j] = w_acc[rr][j];
  }
}

extern "C" void kernel_launch(void* const* d_in, const int* in_sizes, int n_in,
                              void* d_out, int out_size, void* d_ws, size_t ws_size,
                              hipStream_t stream) {
  (void)in_sizes; (void)n_in; (void)d_ws; (void)ws_size; (void)out_size;
  const float* Q = (const float*)d_in[0];
  const float* K = (const float*)d_in[1];
  const float* V = (const float*)d_in[2];
  const unsigned char* mask = (const unsigned char*)d_in[3];  // jax bool -> 1 byte
  // d_in[4] = numb_heads (device scalar); shapes are static, NH hardcoded.
  float* outO = (float*)d_out;
  float* outW = outO + (size_t)BSZ * SQ * EMB;

  dim3 grid(SQ / QT, BSZ);  // (64 q-tiles, 16 batches) = 1024 blocks
  attn_tanhclip_kernel<<<grid, 256, 0, stream>>>(Q, K, V, mask, outO, outW);
}
